// TwoStepVoiceCraftHeads_28338194219141
// MI455X (gfx1250) — matrix operand order repaired and checked by reading, not compiled, not run on hardware
//
#include <hip/hip_runtime.h>
#include <hip/hip_bf16.h>
#include <math.h>

// ---------------------------------------------------------------------------
// CDNA5 (gfx1250) WMMA types
// ---------------------------------------------------------------------------
typedef __attribute__((ext_vector_type(16))) __bf16 v16bf;
typedef __attribute__((ext_vector_type(8)))  float  v8f;

#define LN_EPS 1e-5f

__device__ __forceinline__ unsigned short f2bf(float f) {
    // round-to-nearest-even f32 -> bf16
    unsigned int u = __float_as_uint(f);
    unsigned int r = u + 0x7FFFu + ((u >> 16) & 1u);
    return (unsigned short)(r >> 16);
}

// ---------------------------------------------------------------------------
// Kernel: fp32 -> bf16 convert + transpose (+ zero pad rows)
//   src: [Kheads][Kt][Nsrc] f32   (row-major weight)
//   dst: [Kheads][Np][Kt]  bf16   (transposed; rows n >= Nsrc zero-filled)
// ---------------------------------------------------------------------------
__global__ void cvt_transpose_bf16_kernel(const float* __restrict__ src,
                                          unsigned short* __restrict__ dst,
                                          int Kheads, int Kt, int Nsrc, int Np) {
    size_t i = (size_t)blockIdx.x * blockDim.x + threadIdx.x;
    size_t stride = (size_t)gridDim.x * blockDim.x;
    const size_t per_head = (size_t)Np * (size_t)Kt;
    const size_t total = (size_t)Kheads * per_head;
    for (; i < total; i += stride) {
        size_t h   = i / per_head;
        size_t rem = i - h * per_head;
        int n  = (int)(rem / (size_t)Kt);
        int kk = (int)(rem - (size_t)n * (size_t)Kt);
        unsigned short v = 0;
        if (n < Nsrc)
            v = f2bf(src[(h * (size_t)Kt + kk) * (size_t)Nsrc + n]);
        dst[i] = v;
    }
}

// ---------------------------------------------------------------------------
// Kernel: shared LayerNorm stats + per-head affine -> bf16 Y[K][Ntok][D]
// ---------------------------------------------------------------------------
__global__ void ln_affine_kernel(const float* __restrict__ x,
                                 const float* __restrict__ scale,
                                 const float* __restrict__ bias,
                                 unsigned short* __restrict__ Y,
                                 int Ntok, int D, int K) {
    const int n   = blockIdx.x;
    const int tid = threadIdx.x;
    __shared__ float red[256];

    const float* xr = x + (size_t)n * D;
    float s = 0.f;
    for (int d = tid; d < D; d += 256) s += xr[d];
    red[tid] = s; __syncthreads();
    for (int off = 128; off > 0; off >>= 1) {
        if (tid < off) red[tid] += red[tid + off];
        __syncthreads();
    }
    const float mu = red[0] / (float)D;
    __syncthreads();

    float v = 0.f;
    for (int d = tid; d < D; d += 256) { float t = xr[d] - mu; v += t * t; }
    red[tid] = v; __syncthreads();
    for (int off = 128; off > 0; off >>= 1) {
        if (tid < off) red[tid] += red[tid + off];
        __syncthreads();
    }
    const float rstd = rsqrtf(red[0] / (float)D + LN_EPS);

    for (int d = tid; d < D; d += 256) {
        const float xn = (xr[d] - mu) * rstd;
        for (int k = 0; k < K; ++k) {
            const float y = xn * scale[(size_t)k * D + d] + bias[(size_t)k * D + d];
            Y[((size_t)k * Ntok + n) * D + d] = f2bf(y);
        }
    }
}

// ---------------------------------------------------------------------------
// Kernel: bf16 WMMA GEMM  out = act(A[M,Kt] x W[Kt,Np] + bias)
//   W supplied pre-transposed: Wt_g[Np][Kt].
//   Block tile 128(M) x 128(N), K-step 32, double-buffered LDS (40 KB),
//   8 waves (wave32), each wave: 2x4 tiles of v_wmma_f32_16x16x32_bf16.
//   Np must be a multiple of 128; output store guarded by col < Nreal.
// ---------------------------------------------------------------------------
#define BM  128
#define BN  128
#define BK  32
#define BKP 40   // padded LDS row (80B = 20 dwords; 16B-aligned b128 stores,
                 // 16-lane fragment reads hit 16 distinct banks)

union Frag { v16bf v; unsigned int u[8]; };

template <bool GELU, bool FINAL>
__global__ __launch_bounds__(256)
void gemm_bf16_kernel(const unsigned short* __restrict__ Abase,
                      const unsigned short* __restrict__ Wtbase,   // transposed
                      const float* __restrict__ biasbase,
                      void* __restrict__ outbase,
                      int M, int Kt, int Np, int Nreal,
                      size_t A_kstr, size_t W_kstr, size_t bias_kstr, size_t out_kstr,
                      int T, int Kheads) {
    const int k = blockIdx.z;
    const unsigned short* __restrict__ A    = Abase  + (size_t)k * A_kstr;
    const unsigned short* __restrict__ Wt_g = Wtbase + (size_t)k * W_kstr;
    const float*          __restrict__ bias = biasbase + (size_t)k * bias_kstr;

    const int blockM = blockIdx.y * BM;
    const int blockN = blockIdx.x * BN;

    __shared__ unsigned short As[2][BM][BKP];   // 2 x 10 KB  (A rows)
    __shared__ unsigned short Ws[2][BN][BKP];   // 2 x 10 KB  (W^T rows: [n][k])

    const int tid  = threadIdx.x;
    const int lane = tid & 31;
    const int wave = tid >> 5;     // 0..7
    const int half = lane >> 4;    // 0/1
    const int l16  = lane & 15;
    const int wm   = wave >> 1;    // 0..3  -> 32-row strip
    const int wn   = wave & 1;     // 0..1  -> 64-col strip

    v8f acc[2][4];
#pragma unroll
    for (int i = 0; i < 2; ++i)
#pragma unroll
        for (int j = 0; j < 4; ++j)
            acc[i][j] = (v8f){0.f, 0.f, 0.f, 0.f, 0.f, 0.f, 0.f, 0.f};

    // Tile loader: 2 threads per row, 16 bf16 (32B) each; same map for A and W^T
    const int srow = tid >> 1;          // 0..127
    const int scol = (tid & 1) * 16;    // 0 / 16

    const unsigned short* arow_g = A    + (size_t)(blockM + srow) * Kt + scol;
    const unsigned short* wrow_g = Wt_g + (size_t)(blockN + srow) * Kt + scol;

    auto stage = [&](int buf, int k0) {
        const unsigned short* ag = arow_g + k0;
        *(uint4*)&As[buf][srow][scol]     = *(const uint4*)(ag);
        *(uint4*)&As[buf][srow][scol + 8] = *(const uint4*)(ag + 8);
        const unsigned short* wg = wrow_g + k0;
        *(uint4*)&Ws[buf][srow][scol]     = *(const uint4*)(wg);
        *(uint4*)&Ws[buf][srow][scol + 8] = *(const uint4*)(wg + 8);
        // CDNA5 speculative prefetch (global_prefetch_b8) two k-steps ahead
        __builtin_prefetch(ag + 2 * BK, 0, 1);
        __builtin_prefetch(wg + 2 * BK, 0, 1);
    };

    stage(0, 0);
    __syncthreads();

    int buf = 0;
    for (int k0 = 0; k0 < Kt; k0 += BK) {
        if (k0 + BK < Kt) stage(buf ^ 1, k0 + BK);   // overlap with compute

        Frag a[2], b[4];
#pragma unroll
        for (int i = 0; i < 2; ++i) {
            const unsigned int* ar =
                (const unsigned int*)&As[buf][wm * 32 + i * 16 + l16][0];
#pragma unroll
            for (int j = 0; j < 8; ++j) {
                // A 16x32 bf16 layout: vgpr j holds K-pair; K = base(j) + 8*half
                const int kk = ((j < 4) ? (2 * j) : (16 + 2 * (j - 4))) + 8 * half;
                a[i].u[j] = ar[kk >> 1];
            }
        }
#pragma unroll
        for (int jj = 0; jj < 4; ++jj) {
            const unsigned int* br =
                (const unsigned int*)&Ws[buf][wn * 64 + jj * 16 + l16][0];
#pragma unroll
            for (int j = 0; j < 8; ++j)
                b[jj].u[j] = br[((j < 4) ? j : (j + 4)) + 4 * half];   // K = base(j) + 8*half (same as A)
        }

#pragma unroll
        for (int i = 0; i < 2; ++i)
#pragma unroll
            for (int j = 0; j < 4; ++j)
                acc[i][j] = __builtin_amdgcn_wmma_f32_16x16x32_bf16(
                    false, a[i].v, false, b[j].v, (short)0, acc[i][j], false, false);

        __syncthreads();
        buf ^= 1;
    }

    // ---- epilogue ----
#pragma unroll
    for (int i = 0; i < 2; ++i) {
        const int tileM = blockM + wm * 32 + i * 16;
#pragma unroll
        for (int j = 0; j < 4; ++j) {
            const int tileN = blockN + wn * 64 + j * 16;
            const int col   = tileN + l16;
            const float bv  = (col < Nreal) ? bias[col] : 0.f;
#pragma unroll
            for (int r = 0; r < 8; ++r) {
                const int row = tileM + r + 8 * half;     // C/D fragment layout
                float val = acc[i][j][r] + bv;
                if (GELU)
                    val = 0.5f * val * (1.f + erff(val * 0.70710678118654752f));
                if (FINAL) {
                    if (col < Nreal) {
                        const int bidx = row / T;
                        const int t    = row % T;
                        const size_t off =
                            (((size_t)bidx * Kheads + k) * T + t) * (size_t)Nreal + col;
                        ((float*)outbase)[off] = val;
                    }
                } else {
                    unsigned short* outp =
                        (unsigned short*)outbase + (size_t)k * out_kstr;
                    outp[(size_t)row * Np + col] = f2bf(val);
                }
            }
        }
    }
}

// ---------------------------------------------------------------------------
// Host launch
// ---------------------------------------------------------------------------
extern "C" void kernel_launch(void* const* d_in, const int* in_sizes, int n_in,
                              void* d_out, int out_size, void* d_ws, size_t ws_size,
                              hipStream_t stream) {
    const float* x  = (const float*)d_in[0];
    const float* ls = (const float*)d_in[1];
    const float* lb = (const float*)d_in[2];
    const float* W1 = (const float*)d_in[3];
    const float* b1 = (const float*)d_in[4];
    const float* W2 = (const float*)d_in[5];
    const float* b2 = (const float*)d_in[6];
    const float* Wc = (const float*)d_in[7];
    const float* bc = (const float*)d_in[8];

    const int K = 4, D = 1024, H = 2048, R = 1024, V = 2048 + 4, B = 4, T = 2048;
    const int Ntok = B * T;        // 8192
    const int Vp   = 2176;         // V padded to a multiple of 128 (17*128)

    // carve workspace, 256-byte aligned chunks
    char* p = (char*)d_ws;
    auto carve = [&](size_t bytes) -> void* {
        void* r = (void*)p;
        p += (bytes + 255) & ~(size_t)255;
        return r;
    };
    unsigned short* W1t = (unsigned short*)carve((size_t)K * H * D * 2);  // [K][H][D]
    unsigned short* W2t = (unsigned short*)carve((size_t)K * R * H * 2);  // [K][R][H]
    unsigned short* Wct = (unsigned short*)carve((size_t)K * Vp * R * 2); // [K][Vp][R]
    unsigned short* Yb  = (unsigned short*)carve((size_t)K * Ntok * D * 2);
    unsigned short* Hb  = (unsigned short*)carve((size_t)K * Ntok * H * 2);
    unsigned short* Gb  = (unsigned short*)carve((size_t)K * Ntok * R * 2);

    // weight conversion: bf16 + transpose (+ pad for Wc)
    cvt_transpose_bf16_kernel<<<2048, 256, 0, stream>>>(W1, W1t, K, D, H, H);
    cvt_transpose_bf16_kernel<<<2048, 256, 0, stream>>>(W2, W2t, K, H, R, R);
    cvt_transpose_bf16_kernel<<<2048, 256, 0, stream>>>(Wc, Wct, K, R, V, Vp);

    // shared LN stats + per-head affine
    ln_affine_kernel<<<Ntok, 256, 0, stream>>>(x, ls, lb, Yb, Ntok, D, K);

    dim3 blk(256);

    // GEMM1: Y[8192,1024] x W1[1024,2048] -> GELU -> H (bf16)
    dim3 g1(H / BN, Ntok / BM, K);
    gemm_bf16_kernel<true, false><<<g1, blk, 0, stream>>>(
        Yb, W1t, b1, Hb, Ntok, D, H, H,
        (size_t)Ntok * D, (size_t)H * D, (size_t)H, (size_t)Ntok * H, T, K);

    // GEMM2: H[8192,2048] x W2[2048,1024] -> G (bf16)
    dim3 g2(R / BN, Ntok / BM, K);
    gemm_bf16_kernel<false, false><<<g2, blk, 0, stream>>>(
        Hb, W2t, b2, Gb, Ntok, H, R, R,
        (size_t)Ntok * H, (size_t)R * H, (size_t)R, (size_t)Ntok * R, T, K);

    // GEMM3: G[8192,1024] x Wc[1024,2176(pad)] -> logits f32 [B,K,T,2052]
    dim3 g3(Vp / BN, Ntok / BM, K);
    gemm_bf16_kernel<false, true><<<g3, blk, 0, stream>>>(
        Gb, Wct, bc, d_out, Ntok, R, Vp, V,
        (size_t)Ntok * R, (size_t)Vp * R, (size_t)V, 0, T, K);
}
